// MetaPathAgg_19567871000711
// MI455X (gfx1250) — matrix operand order repaired and checked by reading, not compiled, not run on hardware
//
#include <hip/hip_runtime.h>
#include <hip/hip_bf16.h>

// ---------------- problem constants (match reference) ----------------
constexpr int DD   = 128;
constexpr int KTOT = 6 * DD;     // 768
constexpr int KSTP = KTOT / 32;  // 24 wmma K-steps
constexpr int MT   = 32;         // edges per block (M tile)
constexpr int NB_  = 50000;
constexpr int NBV_ = 100000;
constexpr int NLT_ = 5000;

// workspace layout (in floats)
constexpr long OFF_PV   = 0;                          // pv_pool   [NBV,128]
constexpr long OFF_BVP  = OFF_PV   + (long)NBV_ * DD; // bv_pool   [NBV,128]
constexpr long OFF_BILL = OFF_BVP  + (long)NBV_ * DD; // bill_comm [NB,128]
constexpr long OFF_LTC  = OFF_BILL + (long)NB_  * DD; // lt_comm   [NLT,128]
constexpr long OFF_LTD  = OFF_LTC  + (long)NLT_ * DD; // lt_don    [NLT,128]
constexpr long OFF_LTL  = OFF_LTD  + (long)NLT_ * DD; // lt_lob    [NLT,128]
constexpr long OFF_CPV  = OFF_LTL  + (long)NLT_ * DD;
constexpr long OFF_CR   = OFF_CPV  + NBV_;
constexpr long OFF_CB   = OFF_CR   + NBV_;
constexpr long OFF_CM   = OFF_CB   + NB_;
constexpr long OFF_CD   = OFF_CM   + NLT_;
constexpr long OFF_CL   = OFF_CD   + NLT_;
constexpr long OFF_ZEND = OFF_CL   + NLT_;            // zeroed region end
constexpr long OFF_WPK  = OFF_ZEND;                   // packed W (uint), 24*8*32*8
constexpr int  WPK_N    = KSTP * 8 * 32 * 8;          // 49152 dwords

// ---------------- types for WMMA ----------------
typedef __bf16 bf16_t;
typedef bf16_t v16bf __attribute__((ext_vector_type(16)));
typedef float  v8f   __attribute__((ext_vector_type(8)));
typedef unsigned int v4u __attribute__((ext_vector_type(4)));
typedef unsigned int v2u __attribute__((ext_vector_type(2)));

union Frag { v4u q[2]; v16bf v; };

__device__ __forceinline__ unsigned f2bf(float f) {
    unsigned u = __float_as_uint(f);
    unsigned r = ((u >> 16) & 1u) + 0x7FFFu;   // round-to-nearest-even
    return (u + r) >> 16;
}
__device__ __forceinline__ unsigned pk2bf(float lo, float hi) {
    return f2bf(lo) | (f2bf(hi) << 16);
}

// ---------------- phase 1: scatter-add (sum + count) ----------------
__global__ __launch_bounds__(256)
void scatter_add_kernel(const float* __restrict__ feat,
                        const int* __restrict__ fidx,   // may be null -> identity
                        const int* __restrict__ didx,
                        int nE, float* __restrict__ sum, float* __restrict__ cnt) {
    long i = (long)blockIdx.x * blockDim.x + threadIdx.x;
    int e = (int)(i >> 7);
    int d = (int)(i & 127);
    if (e >= nE) return;
    int s = fidx ? fidx[e] : e;
    int t = didx[e];
    atomicAdd(&sum[(long)t * DD + d], feat[(long)s * DD + d]);
    if (d == 0) atomicAdd(&cnt[t], 1.0f);
}

// ---------------- phase 2: normalize sum / max(cnt,1) ----------------
__global__ __launch_bounds__(256)
void normalize_kernel(float* __restrict__ sum, const float* __restrict__ cnt, int nRows) {
    long i = (long)blockIdx.x * blockDim.x + threadIdx.x;
    if (i >= (long)nRows * DD) return;
    float c = cnt[i >> 7];
    c = c < 1.0f ? 1.0f : c;
    sum[i] = sum[i] / c;
}

// ---------------- phase 3: pack W_fuse[768,128] f32 -> bf16 B-fragment layout ----
// dword p -> (kb, ntile, lane, j): {W[k][col], W[k+1][col]} bf16 pair,
// k = kb*32 + (lane>=16 ? 16 : 0) + 2*j, col = ntile*16 + (lane&15).
__global__ __launch_bounds__(256)
void pack_w_kernel(const float* __restrict__ W, unsigned int* __restrict__ Wpack) {
    int p = blockIdx.x * blockDim.x + threadIdx.x;
    if (p >= WPK_N) return;
    int j     =  p        & 7;
    int lane  = (p >> 3)  & 31;
    int ntile = (p >> 8)  & 7;
    int kb    =  p >> 11;
    int k   = kb * 32 + ((lane >> 4) << 4) + 2 * j;
    int col = ntile * 16 + (lane & 15);
    Wpack[p] = pk2bf(W[k * DD + col], W[(k + 1) * DD + col]);
}

// ---------------- phase 4: gather + WMMA GEMM ----------------
// block = 256 threads = 8 waves; tile = 32 edges x 128 cols; K = 768.
// Each wave owns one 16-col N-tile and two 16-row M-subtiles, reusing its
// B register fragment across both WMMAs per K-step.
__global__ __launch_bounds__(256)
void fuse_gemm_kernel(const float* __restrict__ h_lt,
                      const float* __restrict__ h_topic,
                      const float* __restrict__ b_fuse,
                      const int*   __restrict__ vote_lt,
                      const int*   __restrict__ vote_bv,
                      const int*   __restrict__ bv2bill,
                      const int*   __restrict__ topic_ix,
                      const float* __restrict__ pv_pool,
                      const float* __restrict__ bill_comm,
                      const float* __restrict__ lt_comm,
                      const float* __restrict__ lt_don,
                      const float* __restrict__ lt_lob,
                      const unsigned int* __restrict__ Wpack,
                      float* __restrict__ out, int E_edges) {
    __shared__ unsigned short a_lds[MT * KTOT];   // 48 KB, bf16 bits
    __shared__ int s_lt[MT], s_bv[MT], s_bill[MT], s_top[MT];

    const int ebase_blk = blockIdx.x * MT;

    if (threadIdx.x < MT) {
        int e = ebase_blk + threadIdx.x;
        if (e >= E_edges) e = E_edges - 1;
        int lt = vote_lt[e];
        int bv = vote_bv[e];
        int b  = bv2bill[bv];
        int t  = topic_ix[b];
        if (t < 0) t = 0;
        s_lt[threadIdx.x]   = lt;
        s_bv[threadIdx.x]   = bv;
        s_bill[threadIdx.x] = b;
        s_top[threadIdx.x]  = t;
    }
    __syncthreads();

    // vectorized gather: 32 rows x 192 float4 chunks, f32 -> packed bf16 -> LDS
    constexpr int C4 = KTOT / 4;   // 192 float4 per row
    for (int i = threadIdx.x; i < MT * C4; i += 256) {
        int r  = i / C4;
        int k4 = i - r * C4;
        int seg = k4 >> 5;          // 32 float4 per 128-wide segment
        int d   = (k4 & 31) << 2;   // float offset within segment
        float4 v;
        switch (seg) {
            case 0:  v = *(const float4*)&h_lt     [s_lt[r]   * DD + d]; break;
            case 1:  v = *(const float4*)&pv_pool  [s_bv[r]   * DD + d]; break;
            case 2:  v = *(const float4*)&bill_comm[s_bill[r] * DD + d]; break;
            case 3:  v = *(const float4*)&lt_comm  [s_lt[r]   * DD + d]; break;
            case 4: {
                float4 a = *(const float4*)&lt_don[s_lt[r] * DD + d];
                float4 b = *(const float4*)&lt_lob[s_lt[r] * DD + d];
                v.x = a.x + b.x; v.y = a.y + b.y; v.z = a.z + b.z; v.w = a.w + b.w;
            } break;
            default: v = *(const float4*)&h_topic  [s_top[r]  * DD + d]; break;
        }
        v2u pk; pk.x = pk2bf(v.x, v.y); pk.y = pk2bf(v.z, v.w);
        *reinterpret_cast<v2u*>(&a_lds[r * KTOT + (k4 << 2)]) = pk;   // ds_store_b64
    }
    __syncthreads();

    const int wave = threadIdx.x >> 5;   // N-tile id (0..7)
    const int lane = threadIdx.x & 31;
    const int n    = lane & 15;
    const int half = lane >> 4;
    const int col  = wave * 16 + n;

    v8f c0, c1;
    {
        float bias = b_fuse[col];
        #pragma unroll
        for (int v = 0; v < 8; ++v) { c0[v] = bias; c1[v] = bias; }
    }

    #pragma unroll 4
    for (int kb = 0; kb < KSTP; ++kb) {
        // B fragment: pre-packed, 8 contiguous dwords per lane; reused twice
        Frag fb;
        const unsigned int* wp = Wpack + (((kb * 8 + wave) * 32 + lane) << 3);
        fb.q[0] = *reinterpret_cast<const v4u*>(wp);
        fb.q[1] = *reinterpret_cast<const v4u*>(wp + 4);

        // A fragments: lanes 0-15 -> K {0..15}; lanes 16-31 -> K {16..31}
        int kofs = kb * 32 + half * 16;
        Frag fa0, fa1;
        int a0 = n * KTOT + kofs;             // rows 0..15
        int a1 = (16 + n) * KTOT + kofs;      // rows 16..31
        fa0.q[0] = *reinterpret_cast<const v4u*>(&a_lds[a0]);
        fa0.q[1] = *reinterpret_cast<const v4u*>(&a_lds[a0 + 8]);
        fa1.q[0] = *reinterpret_cast<const v4u*>(&a_lds[a1]);
        fa1.q[1] = *reinterpret_cast<const v4u*>(&a_lds[a1 + 8]);

        c0 = __builtin_amdgcn_wmma_f32_16x16x32_bf16(
                false, fa0.v, false, fb.v, (short)0, c0, false, false);
        c1 = __builtin_amdgcn_wmma_f32_16x16x32_bf16(
                false, fa1.v, false, fb.v, (short)0, c1, false, false);
    }

    // epilogue: int-only addressing; whole-tile fast path (E % 32 == 0 -> always)
    int row0 = ebase_blk + half * 8;          // M sub-tile 0
    int row1 = row0 + 16;                     // M sub-tile 1
    if (ebase_blk + MT <= E_edges) {
        #pragma unroll
        for (int v = 0; v < 8; ++v) out[(row0 + v) * DD + col] = c0[v];
        #pragma unroll
        for (int v = 0; v < 8; ++v) out[(row1 + v) * DD + col] = c1[v];
    } else {
        #pragma unroll
        for (int v = 0; v < 8; ++v) if (row0 + v < E_edges) out[(row0 + v) * DD + col] = c0[v];
        #pragma unroll
        for (int v = 0; v < 8; ++v) if (row1 + v < E_edges) out[(row1 + v) * DD + col] = c1[v];
    }
}

// ---------------- host-side launch ----------------
extern "C" void kernel_launch(void* const* d_in, const int* in_sizes, int n_in,
                              void* d_out, int out_size, void* d_ws, size_t ws_size,
                              hipStream_t stream) {
    const float* h_bv    = (const float*)d_in[1];
    const float* h_lt    = (const float*)d_in[2];
    const float* h_comm  = (const float*)d_in[3];
    const float* h_donor = (const float*)d_in[4];
    const float* h_lobby = (const float*)d_in[5];
    const float* h_topic = (const float*)d_in[6];
    const float* W_fuse  = (const float*)d_in[7];
    const float* b_fuse  = (const float*)d_in[8];
    const int* vote_lt = (const int*)d_in[9];
    const int* vote_bv = (const int*)d_in[10];
    const int* bv2bill = (const int*)d_in[11];
    const int* topic_ix= (const int*)d_in[12];
    const int* pv_src  = (const int*)d_in[13];
    const int* pv_dst  = (const int*)d_in[14];
    const int* r_src   = (const int*)d_in[15];
    const int* r_dst   = (const int*)d_in[16];
    const int* m_src   = (const int*)d_in[17];
    const int* m_dst   = (const int*)d_in[18];
    const int* don_src = (const int*)d_in[19];
    const int* don_dst = (const int*)d_in[20];
    const int* lob_src = (const int*)d_in[21];
    const int* lob_dst = (const int*)d_in[22];

    const int E    = in_sizes[9];    // 300000
    const int EPV  = in_sizes[13];
    const int ERD  = in_sizes[15];
    const int EM   = in_sizes[17];
    const int EDON = in_sizes[19];
    const int ELOB = in_sizes[21];

    float* ws = (float*)d_ws;
    float* pv_pool   = ws + OFF_PV;
    float* bv_pool   = ws + OFF_BVP;
    float* bill_comm = ws + OFF_BILL;
    float* lt_comm   = ws + OFF_LTC;
    float* lt_don    = ws + OFF_LTD;
    float* lt_lob    = ws + OFF_LTL;
    float* c_pv  = ws + OFF_CPV;
    float* c_r   = ws + OFF_CR;
    float* c_b   = ws + OFF_CB;
    float* c_m   = ws + OFF_CM;
    float* c_d   = ws + OFF_CD;
    float* c_l   = ws + OFF_CL;
    unsigned int* Wpack = (unsigned int*)(ws + OFF_WPK);

    hipMemsetAsync(d_ws, 0, (size_t)OFF_ZEND * sizeof(float), stream);

    auto gblk = [](long n) { return (unsigned)((n + 255) / 256); };

    scatter_add_kernel<<<gblk((long)EPV  * DD), 256, 0, stream>>>(h_bv,    pv_src, pv_dst, EPV,  pv_pool, c_pv);
    scatter_add_kernel<<<gblk((long)ERD  * DD), 256, 0, stream>>>(h_comm,  r_dst,  r_src,  ERD,  bv_pool, c_r);
    scatter_add_kernel<<<gblk((long)EM   * DD), 256, 0, stream>>>(h_comm,  m_dst,  m_src,  EM,   lt_comm, c_m);
    scatter_add_kernel<<<gblk((long)EDON * DD), 256, 0, stream>>>(h_donor, don_src, don_dst, EDON, lt_don, c_d);
    scatter_add_kernel<<<gblk((long)ELOB * DD), 256, 0, stream>>>(h_lobby, lob_src, lob_dst, ELOB, lt_lob, c_l);

    normalize_kernel<<<gblk((long)NBV_ * DD), 256, 0, stream>>>(pv_pool, c_pv, NBV_);
    normalize_kernel<<<gblk((long)NBV_ * DD), 256, 0, stream>>>(bv_pool, c_r,  NBV_);
    normalize_kernel<<<gblk((long)NLT_ * DD), 256, 0, stream>>>(lt_comm, c_m,  NLT_);
    normalize_kernel<<<gblk((long)NLT_ * DD), 256, 0, stream>>>(lt_don,  c_d,  NLT_);
    normalize_kernel<<<gblk((long)NLT_ * DD), 256, 0, stream>>>(lt_lob,  c_l,  NLT_);

    scatter_add_kernel<<<gblk((long)NBV_ * DD), 256, 0, stream>>>(bv_pool, nullptr, bv2bill, NBV_, bill_comm, c_b);
    normalize_kernel<<<gblk((long)NB_ * DD), 256, 0, stream>>>(bill_comm, c_b, NB_);

    pack_w_kernel<<<gblk(WPK_N), 256, 0, stream>>>(W_fuse, Wpack);

    unsigned nblk = (unsigned)((E + MT - 1) / MT);
    fuse_gemm_kernel<<<nblk, 256, 0, stream>>>(
        h_lt, h_topic, b_fuse, vote_lt, vote_bv, bv2bill, topic_ix,
        pv_pool, bill_comm, lt_comm, lt_don, lt_lob, Wpack,
        (float*)d_out, E);
}
